// AttentionBlock_64905545777197
// MI455X (gfx1250) — matrix operand order repaired and checked by reading, not compiled, not run on hardware
//
#include <hip/hip_runtime.h>

typedef __attribute__((ext_vector_type(16))) _Float16 v16h;
typedef __attribute__((ext_vector_type(8)))  _Float16 v8h;
typedef __attribute__((ext_vector_type(8)))  float    v8f;

#define B_  8
#define C_  512
#define N_  4096
#define G_  32

#define QK_STRIDE 520   // 512 halves + 8 pad (bank spread for row-parallel b128 reads)
#define VK_STRIDE 40    // 32 halves + 8 pad
#define GT_STRIDE 72    // 64 halves + 8 pad

// ---------------------------------------------------------------------------
// WMMA fragment loads: contiguous 2 x b128 per lane.
// A (16xK=32, M=row):  element i -> K = 8*half + i   (i<8),  16+8*half+(i-8)
// B (K=32xN=16, N=row of B^T storage): element i -> K = 8*half + i (i<8), 16+8*half+(i-8)
// ---------------------------------------------------------------------------
union U16x16 { v16h v; uint4 u[2]; };

static __device__ __forceinline__ v16h fragA(const _Float16* rowp, int half) {
    U16x16 t;
    t.u[0] = *(const uint4*)(rowp + 8 * half);
    t.u[1] = *(const uint4*)(rowp + 16 + 8 * half);
    return t.v;
}
static __device__ __forceinline__ v16h fragB(const _Float16* rowp, int half) {
    U16x16 t;
    t.u[0] = *(const uint4*)(rowp + 8 * half);
    t.u[1] = *(const uint4*)(rowp + 16 + 8 * half);
    return t.v;
}
union F32x8 { float4 f[2]; float a[8]; };

// ---------------------------------------------------------------------------
// Tensor Data Mover staging (guarded; manual-copy fallback always compiles)
// ---------------------------------------------------------------------------
#if defined(__gfx1250__) && __has_builtin(__builtin_amdgcn_s_wait_tensorcnt) && \
    (__has_builtin(__builtin_amdgcn_tensor_load_to_lds) || \
     __has_builtin(__builtin_amdgcn_tensor_load_to_lds_d2))
#define USE_TDM 1
#else
#define USE_TDM 0
#endif

// Async global->LDS copies (ASYNCcnt path), guarded.
#if defined(__gfx1250__) && \
    __has_builtin(__builtin_amdgcn_global_load_async_to_lds_b128) && \
    __has_builtin(__builtin_amdgcn_s_wait_asynccnt)
#define USE_ASYNC 1
#else
#define USE_ASYNC 0
#endif

#if USE_TDM || USE_ASYNC
typedef unsigned v4u __attribute__((ext_vector_type(4)));
typedef int      v8i __attribute__((ext_vector_type(8)));
typedef int      v4i __attribute__((ext_vector_type(4)));
#endif

#if USE_TDM
// 2-D tile load: tensor (dim0 contiguous, dim1 rows, stride s0), tile tl0 x tl1,
// LDS padding: (padamt+1) dwords after every 2^(padint+1) dwords.
static __device__ __forceinline__ void tdm_load_2d(unsigned lds, const void* g,
                                                   unsigned td0, unsigned td1,
                                                   unsigned tl0, unsigned tl1,
                                                   unsigned long long s0,
                                                   unsigned padint, unsigned padamt) {
    unsigned long long ga = (unsigned long long)g;
    unsigned d0 = 1u;                                           // count=1 user D#
    unsigned d1 = lds;                                          // lds_addr
    unsigned d2 = (unsigned)(ga & 0xffffffffu);                 // global_addr lo
    unsigned d3 = (unsigned)((ga >> 32) & 0x1ffffffu) | (2u << 30);  // hi | type=2
    v8i g1 = { (int)((1u << 16) | (1u << 20) | (padint << 22) | (padamt << 25)),
               (int)((td0 & 0xffffu) << 16),
               (int)(((td0 >> 16) & 0xffffu) | ((td1 & 0xffffu) << 16)),
               (int)(((td1 >> 16) & 0xffffu) | (tl0 << 16)),
               (int)(tl1 & 0xffffu),
               (int)(s0 & 0xffffffffull),
               (int)((s0 >> 32) & 0xffffu),
               0 };
#if __has_builtin(__builtin_amdgcn_tensor_load_to_lds_d2)
    v4i g0 = { (int)d0, (int)d1, (int)d2, (int)d3 };
    __builtin_amdgcn_tensor_load_to_lds_d2(g0, g1, 0);
#elif defined(__clang_major__) && (__clang_major__ >= 23)
    v4u g0 = { d0, d1, d2, d3 };
    v4i z4 = { 0, 0, 0, 0 };
    v8i z8 = { 0, 0, 0, 0, 0, 0, 0, 0 };
    __builtin_amdgcn_tensor_load_to_lds(g0, g1, z4, z4, z8, 0);
#else
    v4u g0 = { d0, d1, d2, d3 };
    v4i z4 = { 0, 0, 0, 0 };
    __builtin_amdgcn_tensor_load_to_lds(g0, g1, z4, z4, 0);
#endif
}
static __device__ __forceinline__ unsigned ldsLow32(const void* p) {
    return (unsigned)(unsigned long long)p;   // flat LDS addr low bits = LDS offset
}
#endif

#if USE_ASYNC
// Builtin signature (from hipcc diagnostic): param0 = int4 addrspace(1)* (global),
// param1 expected int4 addrspace(3)* (LDS); build exact types via integer casts.
typedef v4i __attribute__((address_space(1)))* as1v4p;
typedef v4i __attribute__((address_space(3)))* as3v4p;
static __device__ __forceinline__ void async_cp_b128(void* lds, const void* g) {
    __builtin_amdgcn_global_load_async_to_lds_b128(
        (as1v4p)(unsigned long long)g,
        (as3v4p)(unsigned)(unsigned long long)lds, 0, 0);
}
#endif

// ---------------------------------------------------------------------------
// fp32 -> fp16 weight conversion
// ---------------------------------------------------------------------------
__global__ __launch_bounds__(256) void cvt_f16_kernel(const float* __restrict__ s,
                                                      _Float16* __restrict__ d, int n) {
    int i = blockIdx.x * 256 + threadIdx.x;
    if (i < n) d[i] = (_Float16)s[i];
}

// ---------------------------------------------------------------------------
// GroupNorm statistics: one block per (batch,group)
// ---------------------------------------------------------------------------
__global__ __launch_bounds__(256) void gn_stats_kernel(const float* __restrict__ x,
                                                       float* __restrict__ mu,
                                                       float* __restrict__ rstd) {
    __shared__ float s1[256], s2[256];
    const int tid = threadIdx.x;
    const float4* xv = (const float4*)(x + (size_t)blockIdx.x * 65536);
    float a = 0.f, b = 0.f;
    for (int i = tid; i < 16384; i += 256) {
        float4 v = xv[i];
        a += v.x + v.y + v.z + v.w;
        b += v.x * v.x + v.y * v.y + v.z * v.z + v.w * v.w;
    }
    s1[tid] = a; s2[tid] = b;
    __syncthreads();
    for (int off = 128; off > 0; off >>= 1) {
        if (tid < off) { s1[tid] += s1[tid + off]; s2[tid] += s2[tid + off]; }
        __syncthreads();
    }
    if (tid == 0) {
        float m   = s1[0] * (1.f / 65536.f);
        float var = s2[0] * (1.f / 65536.f) - m * m;
        mu[blockIdx.x]   = m;
        rstd[blockIdx.x] = rsqrtf(var + 1e-6f);
    }
}

// ---------------------------------------------------------------------------
// GroupNorm apply + transpose: x (b,c,n) f32 -> h (b,n,c) f16 token-major.
// ---------------------------------------------------------------------------
__global__ __launch_bounds__(256) void gn_apply_kernel(const float* __restrict__ x,
                                                       const float* __restrict__ mu,
                                                       const float* __restrict__ rstd,
                                                       const float* __restrict__ sc,
                                                       const float* __restrict__ bi,
                                                       _Float16* __restrict__ ht) {
    __shared__ __align__(16) _Float16 lt[64 * GT_STRIDE];
    const int tid  = threadIdx.x;
    const int nq   = tid & 15;
    const int tc   = tid >> 4;
    const int nBlk = blockIdx.x * 64;
    const int cBlk = blockIdx.y * 64;
    const int b    = blockIdx.z;
    #pragma unroll
    for (int p = 0; p < 4; ++p) {
        int cl = p * 16 + tc;
        int c  = cBlk + cl;
        int bg = b * G_ + (c >> 4);
        float m = mu[bg], r = rstd[bg];
        float s = sc[c] * r;
        float t = bi[c] - m * s;
        float4 v = *(const float4*)&x[((size_t)b * C_ + c) * N_ + nBlk + nq * 4];
        lt[(nq * 4 + 0) * GT_STRIDE + cl] = (_Float16)(v.x * s + t);
        lt[(nq * 4 + 1) * GT_STRIDE + cl] = (_Float16)(v.y * s + t);
        lt[(nq * 4 + 2) * GT_STRIDE + cl] = (_Float16)(v.z * s + t);
        lt[(nq * 4 + 3) * GT_STRIDE + cl] = (_Float16)(v.w * s + t);
    }
    __syncthreads();
    for (int it = tid; it < 512; it += 256) {
        int row = it >> 3, ch = it & 7;
        *(uint4*)&ht[((size_t)b * N_ + nBlk + row) * C_ + cBlk + ch * 8] =
            *(const uint4*)&lt[row * GT_STRIDE + ch * 8];
    }
}

// ---------------------------------------------------------------------------
// 1x1 conv GEMM: D[o,n] = sum_c W[o,c] * Ht[n,c] + bias[o]
// Block tile 128o x 64n, 8 waves (4x2), each wave 32x32 = 4 wmma per K-step.
// ---------------------------------------------------------------------------
template <int OUT>
__global__ __launch_bounds__(256) void proj_kernel(const _Float16* __restrict__ Wf,
                                                   const float* __restrict__ bias,
                                                   const _Float16* __restrict__ Hin,
                                                   _Float16* __restrict__ Yout,
                                                   const float* __restrict__ Xres,
                                                   float* __restrict__ Fout) {
    __shared__ __align__(16) _Float16 lwA[128 * VK_STRIDE];  // W tile  [o][c]
    __shared__ __align__(16) _Float16 lhB[64 * VK_STRIDE];   // H tile  [n][c]

    const int tid  = threadIdx.x;
    const int lane = tid & 31, wave = tid >> 5;
    const int half = lane >> 4, l16 = lane & 15;
    const int waveO = wave >> 1, waveN = wave & 1;
    const int nBlk = blockIdx.x * 64;
    const int oBlk = blockIdx.y * 128;
    const int b    = blockIdx.z;
    const _Float16* Hb = Hin + (size_t)b * N_ * C_;

    v8f acc[2][2] = {};
    for (int kc = 0; kc < C_; kc += 32) {
        __syncthreads();
#if USE_ASYNC
        for (int it = tid; it < 512; it += 256) {
            int row = it >> 2, ch = it & 3;
            async_cp_b128(&lwA[row * VK_STRIDE + ch * 8],
                          &Wf[(size_t)(oBlk + row) * C_ + kc + ch * 8]);
        }
        {
            int row = tid >> 2, ch = tid & 3;
            async_cp_b128(&lhB[row * VK_STRIDE + ch * 8],
                          &Hb[(size_t)(nBlk + row) * C_ + kc + ch * 8]);
        }
        __builtin_amdgcn_s_wait_asynccnt(0);
#else
        for (int it = tid; it < 512; it += 256) {           // 128 rows x 4 x 16B
            int row = it >> 2, ch = it & 3;
            *(uint4*)&lwA[row * VK_STRIDE + ch * 8] =
                *(const uint4*)&Wf[(size_t)(oBlk + row) * C_ + kc + ch * 8];
        }
        {
            int row = tid >> 2, ch = tid & 3;               // 64 rows x 4 x 16B
            *(uint4*)&lhB[row * VK_STRIDE + ch * 8] =
                *(const uint4*)&Hb[(size_t)(nBlk + row) * C_ + kc + ch * 8];
        }
        if (kc + 32 < C_)
            __builtin_prefetch(&Hb[(size_t)(nBlk + (tid & 63)) * C_ + kc + 32], 0, 0);
#endif
        __syncthreads();

        v16h a0 = fragA(&lwA[(waveO * 32 + l16) * VK_STRIDE], half);
        v16h a1 = fragA(&lwA[(waveO * 32 + 16 + l16) * VK_STRIDE], half);
        v16h b0 = fragB(&lhB[(waveN * 32 + l16) * VK_STRIDE], half);
        v16h b1 = fragB(&lhB[(waveN * 32 + 16 + l16) * VK_STRIDE], half);
        acc[0][0] = __builtin_amdgcn_wmma_f32_16x16x32_f16(false, a0, false, b0, (short)0, acc[0][0], false, false);
        acc[0][1] = __builtin_amdgcn_wmma_f32_16x16x32_f16(false, a0, false, b1, (short)0, acc[0][1], false, false);
        acc[1][0] = __builtin_amdgcn_wmma_f32_16x16x32_f16(false, a1, false, b0, (short)0, acc[1][0], false, false);
        acc[1][1] = __builtin_amdgcn_wmma_f32_16x16x32_f16(false, a1, false, b1, (short)0, acc[1][1], false, false);
    }

    #pragma unroll
    for (int ot = 0; ot < 2; ++ot) {
        const int ob = oBlk + waveO * 32 + ot * 16 + 8 * half;   // 8 consecutive o
        F32x8 bv;
        bv.f[0] = *(const float4*)&bias[ob];
        bv.f[1] = *(const float4*)&bias[ob + 4];
        #pragma unroll
        for (int nt = 0; nt < 2; ++nt) {
            const int n = nBlk + waveN * 32 + nt * 16 + l16;
            if (OUT == 0) {                       // token-major f16, packed store
                v8h pk;
                #pragma unroll
                for (int r = 0; r < 8; ++r)
                    pk[r] = (_Float16)(acc[ot][nt][r] + bv.a[r]);
                *(v8h*)&Yout[((size_t)b * N_ + n) * C_ + ob] = pk;
            } else if (OUT == 1) {                // channel-major f16
                #pragma unroll
                for (int r = 0; r < 8; ++r)
                    Yout[((size_t)b * C_ + ob + r) * N_ + n] =
                        (_Float16)(acc[ot][nt][r] + bv.a[r]);
            } else {                              // channel-major f32 + residual
                #pragma unroll
                for (int r = 0; r < 8; ++r) {
                    size_t idx = ((size_t)b * C_ + ob + r) * N_ + n;
                    Fout[idx] = Xres[idx] + acc[ot][nt][r] + bv.a[r];
                }
            }
        }
    }
}

// ---------------------------------------------------------------------------
// Flash attention: 16-query tile per block, 32-key tiles; 8 waves.
// Qt/Kt token-major (n,c); V channel-major (c,n); output token-major.
// ---------------------------------------------------------------------------
__global__ __launch_bounds__(256) void attn_kernel(const _Float16* __restrict__ Q,
                                                   const _Float16* __restrict__ K,
                                                   const _Float16* __restrict__ V,
                                                   _Float16* __restrict__ O) {
    __shared__ __align__(16) _Float16 qs[16 * QK_STRIDE];   // [q][c]
    __shared__ __align__(16) _Float16 ks[32 * QK_STRIDE];   // [k][c]
    __shared__ __align__(16) _Float16 vs[C_ * VK_STRIDE];   // [c][k]
    __shared__ __align__(16) float    sS[16 * 32];          // [q][k]
    __shared__ __align__(16) _Float16 sP[16 * VK_STRIDE];   // [q][k]
    __shared__ __align__(16) float    sM[16], sL[16], sA[16];

    const int tid  = threadIdx.x;
    const int lane = tid & 31, wave = tid >> 5;
    const int half = lane >> 4, l16 = lane & 15;
    const int qBase = blockIdx.x * 16;
    const int b     = blockIdx.y;
    const _Float16* Qb = Q + (size_t)b * N_ * C_;   // token-major
    const _Float16* Kb = K + (size_t)b * N_ * C_;   // token-major
    const _Float16* Vb = V + (size_t)b * C_ * N_;   // channel-major

    // ---- stage the 16x512 query tile once ----
#if USE_TDM
    if (wave == 0) {
        tdm_load_2d(ldsLow32(qs), Qb + (size_t)qBase * C_, C_, N_, C_, 16,
                    (unsigned long long)C_, 7u, 3u);        // pad 4dw/256dw -> 520
        __builtin_amdgcn_s_wait_tensorcnt(0);
    }
#elif USE_ASYNC
    for (int it = tid; it < 1024; it += 256) {
        int row = it >> 6, ch = it & 63;
        async_cp_b128(&qs[row * QK_STRIDE + ch * 8],
                      &Qb[(size_t)(qBase + row) * C_ + ch * 8]);
    }
    __builtin_amdgcn_s_wait_asynccnt(0);
#else
    for (int it = tid; it < 1024; it += 256) {              // 16 rows x 64 x 16B
        int row = it >> 6, ch = it & 63;
        *(uint4*)&qs[row * QK_STRIDE + ch * 8] =
            *(const uint4*)&Qb[(size_t)(qBase + row) * C_ + ch * 8];
    }
#endif
    if (tid < 16) { sM[tid] = -3.0e38f; sL[tid] = 0.f; }
    __syncthreads();

    const int cWave = wave << 6;
    const float scale = 0.044194173824159216f;   // 512^-0.5
    v8f acc[4] = {};

    for (int kt = 0; kt < N_ / 32; ++kt) {
        const int kBase = kt * 32;
#if USE_TDM
        if (wave == 0) {
            tdm_load_2d(ldsLow32(ks), Kb + (size_t)kBase * C_, C_, N_, C_, 32,
                        (unsigned long long)C_, 7u, 3u);    // [k][c], stride 520
            tdm_load_2d(ldsLow32(vs), Vb + kBase, N_, C_, 32, C_,
                        (unsigned long long)N_, 3u, 3u);    // [c][k], stride 40
            __builtin_amdgcn_s_wait_tensorcnt(0);
        }
#elif USE_ASYNC
        for (int it = tid; it < 2048; it += 256) {
            int row = it >> 6, ch = it & 63;
            async_cp_b128(&ks[row * QK_STRIDE + ch * 8],
                          &Kb[(size_t)(kBase + row) * C_ + ch * 8]);
        }
        for (int it = tid; it < 2048; it += 256) {
            int row = it >> 2, ch = it & 3;
            async_cp_b128(&vs[row * VK_STRIDE + ch * 8],
                          &Vb[(size_t)row * N_ + kBase + ch * 8]);
        }
        __builtin_amdgcn_s_wait_asynccnt(0);
#else
        for (int it = tid; it < 2048; it += 256) {          // 32 rows x 64 x 16B
            int row = it >> 6, ch = it & 63;
            *(uint4*)&ks[row * QK_STRIDE + ch * 8] =
                *(const uint4*)&Kb[(size_t)(kBase + row) * C_ + ch * 8];
        }
        for (int it = tid; it < 2048; it += 256) {          // 512 rows x 4 x 16B
            int row = it >> 2, ch = it & 3;
            *(uint4*)&vs[row * VK_STRIDE + ch * 8] =
                *(const uint4*)&Vb[(size_t)row * N_ + kBase + ch * 8];
        }
#endif
        sS[tid] = 0.f; sS[tid + 256] = 0.f;
        __syncthreads();

        // --- partial S = Qt x K^T over this wave's 64 channels (4 wmma) ---
        v8f s0 = {}, s1 = {};
        #pragma unroll
        for (int cc = 0; cc < 64; cc += 32) {
            int cb = cWave + cc;
            v16h aq = fragA(&qs[l16 * QK_STRIDE + cb], half);
            v16h b0 = fragB(&ks[l16 * QK_STRIDE + cb], half);
            v16h b1 = fragB(&ks[(16 + l16) * QK_STRIDE + cb], half);
            s0 = __builtin_amdgcn_wmma_f32_16x16x32_f16(false, aq, false, b0, (short)0, s0, false, false);
            s1 = __builtin_amdgcn_wmma_f32_16x16x32_f16(false, aq, false, b1, (short)0, s1, false, false);
        }
        #pragma unroll
        for (int r = 0; r < 8; ++r) {            // cross-wave channel reduce
            int q = r + (half << 3);
            atomicAdd(&sS[q * 32 + l16],      s0[r]);   // ds_add_f32
            atomicAdd(&sS[q * 32 + 16 + l16], s1[r]);
        }
        __syncthreads();

        // --- online softmax, one thread per query row ---
        if (tid < 16) {
            const float4* rs = (const float4*)&sS[tid * 32];
            float mOld = sM[tid], mNew = mOld;
            #pragma unroll
            for (int j = 0; j < 8; ++j) {
                float4 v = rs[j];
                mNew = fmaxf(mNew, fmaxf(fmaxf(v.x, v.y), fmaxf(v.z, v.w)) * scale);
            }
            float alpha = __expf(mOld - mNew);
            float lsum = 0.f;
            #pragma unroll
            for (int j = 0; j < 8; ++j) {
                float4 v = rs[j];
                float p0 = __expf(v.x * scale - mNew), p1 = __expf(v.y * scale - mNew);
                float p2 = __expf(v.z * scale - mNew), p3 = __expf(v.w * scale - mNew);
                lsum += p0 + p1 + p2 + p3;
                sP[tid * VK_STRIDE + j * 4 + 0] = (_Float16)p0;
                sP[tid * VK_STRIDE + j * 4 + 1] = (_Float16)p1;
                sP[tid * VK_STRIDE + j * 4 + 2] = (_Float16)p2;
                sP[tid * VK_STRIDE + j * 4 + 3] = (_Float16)p3;
            }
            sM[tid] = mNew;
            sL[tid] = sL[tid] * alpha + lsum;
            sA[tid] = alpha;
        }
        __syncthreads();

        // --- O[q,c] += P x V^T over this wave's 64 channels (4 wmma) ---
        v16h ap = fragA(&sP[l16 * VK_STRIDE], half);    // A = P (q x k)
        F32x8 av;
        av.f[0] = *(const float4*)&sA[8 * half];
        av.f[1] = *(const float4*)&sA[8 * half + 4];
        #pragma unroll
        for (int m = 0; m < 4; ++m) {
            v16h bv = fragB(&vs[(cWave + m * 16 + l16) * VK_STRIDE], half);
            v8f t = acc[m];
            #pragma unroll
            for (int r = 0; r < 8; ++r) t[r] *= av.a[r];
            acc[m] = __builtin_amdgcn_wmma_f32_16x16x32_f16(false, ap, false, bv, (short)0, t, false, false);
        }
        __syncthreads();
    }

    // epilogue: divide by l, write token-major output
    F32x8 lv;
    lv.f[0] = *(const float4*)&sL[8 * half];
    lv.f[1] = *(const float4*)&sL[8 * half + 4];
    #pragma unroll
    for (int m = 0; m < 4; ++m) {
        #pragma unroll
        for (int r = 0; r < 8; ++r) {
            int q = qBase + r + 8 * half;
            int c = cWave + m * 16 + l16;
            O[((size_t)b * N_ + q) * C_ + c] = (_Float16)(acc[m][r] / lv.a[r]);
        }
    }
}

// ---------------------------------------------------------------------------
extern "C" void kernel_launch(void* const* d_in, const int* in_sizes, int n_in,
                              void* d_out, int out_size, void* d_ws, size_t ws_size,
                              hipStream_t stream) {
    const float* x   = (const float*)d_in[0];
    const float* gns = (const float*)d_in[1];
    const float* gnb = (const float*)d_in[2];
    const float* Wq  = (const float*)d_in[3];
    const float* bq  = (const float*)d_in[4];
    const float* Wk  = (const float*)d_in[5];
    const float* bk  = (const float*)d_in[6];
    const float* Wv  = (const float*)d_in[7];
    const float* bv  = (const float*)d_in[8];
    const float* Wo  = (const float*)d_in[9];
    const float* bo  = (const float*)d_in[10];
    float* out = (float*)d_out;

    const size_t WBYTES = (size_t)C_ * C_ * sizeof(_Float16);
    const size_t HBYTES = (size_t)B_ * C_ * N_ * sizeof(_Float16);

    char* w = (char*)d_ws;
    float*    mu   = (float*)w;     w += 1024;
    float*    rstd = (float*)w;     w += 1024;
    _Float16* Wfq  = (_Float16*)w;  w += WBYTES;
    _Float16* Wfk  = (_Float16*)w;  w += WBYTES;
    _Float16* Wfv  = (_Float16*)w;  w += WBYTES;
    _Float16* Wfo  = (_Float16*)w;  w += WBYTES;
    _Float16* h16  = (_Float16*)w;  w += HBYTES;   // token-major (b,n,c)
    _Float16* q16  = (_Float16*)w;  w += HBYTES;   // token-major
    _Float16* k16  = (_Float16*)w;  w += HBYTES;   // token-major
    _Float16* v16  = (_Float16*)w;  w += HBYTES;   // channel-major (b,c,n)
    _Float16* a16  = (_Float16*)w;  w += HBYTES;   // token-major

    const int WN = C_ * C_;
    cvt_f16_kernel<<<WN / 256, 256, 0, stream>>>(Wq, Wfq, WN);
    cvt_f16_kernel<<<WN / 256, 256, 0, stream>>>(Wk, Wfk, WN);
    cvt_f16_kernel<<<WN / 256, 256, 0, stream>>>(Wv, Wfv, WN);
    cvt_f16_kernel<<<WN / 256, 256, 0, stream>>>(Wo, Wfo, WN);

    gn_stats_kernel<<<B_ * G_, 256, 0, stream>>>(x, mu, rstd);
    dim3 gg(N_ / 64, C_ / 64, B_);
    gn_apply_kernel<<<gg, 256, 0, stream>>>(x, mu, rstd, gns, gnb, h16);

    dim3 pg(N_ / 64, C_ / 128, B_);
    proj_kernel<0><<<pg, 256, 0, stream>>>(Wfq, bq, h16, q16, nullptr, nullptr);
    proj_kernel<0><<<pg, 256, 0, stream>>>(Wfk, bk, h16, k16, nullptr, nullptr);
    proj_kernel<1><<<pg, 256, 0, stream>>>(Wfv, bv, h16, v16, nullptr, nullptr);

    dim3 ag(N_ / 16, B_);
    attn_kernel<<<ag, 256, 0, stream>>>(q16, k16, v16, a16);

    proj_kernel<2><<<pg, 256, 0, stream>>>(Wfo, bo, a16, nullptr, x, out);
}
